// ET_33895881900641
// MI455X (gfx1250) — compile-verified, matrix operand order checked by reading, not run on hardware
//
#include <hip/hip_runtime.h>

// ---------------------------------------------------------------------------
// Energy Transformer forward on MI455X (gfx1250, wave32, bf16 WMMA).
// GEMMs: 128x128 block tile, 8 waves, K-step 32, double-buffered fragment-
// major LDS (every WMMA operand = one 32B LDS read), staged with
// GLOBAL_LOAD_ASYNC_TO_LDS_B128 when the toolchain exposes it.
// Weights are pre-swizzled in global memory into WMMA B-fragment order.
// ---------------------------------------------------------------------------

typedef __bf16 bf16;
typedef __attribute__((ext_vector_type(16))) __bf16 v16bf;
typedef __attribute__((ext_vector_type(8)))  __bf16 v8bf;
typedef __attribute__((ext_vector_type(8)))  float  v8f;
typedef __attribute__((ext_vector_type(4)))  int    v4i;

#define DEVI __device__ __forceinline__

#if __has_builtin(__builtin_amdgcn_global_load_async_to_lds_b128)
#define ET_ASYNC 1
#else
#define ET_ASYNC 0
#endif

// Problem constants (match the reference)
constexpr int  BATCH   = 64;
constexpr int  S_TOK   = 197;    // 196 patches + cls
constexpr int  S_PAD   = 208;    // padded to 13 * 16
constexpr int  TKN     = 768;
constexpr int  QKD     = 64;
constexpr int  HEADS   = 12;
constexpr int  HID     = 3072;
constexpr int  NCLS    = 1000;
constexpr int  NCLS_P  = 1008;   // padded to 63 * 16
constexpr int  K_STEPS = 12;
constexpr long MT      = (long)BATCH * S_PAD;   // 13312 token rows (padded)
constexpr float ATTN_BETA = 0.125f;             // 1/sqrt(64)
constexpr float LN_EPS    = 1e-5f;

// Epilogue flags
constexpr int E_F32  = 1;   // write f32
constexpr int E_BF16 = 2;   // write bf16
constexpr int E_ACC  = 4;   // accumulate into f32
constexpr int E_BIAS = 8;   // + bias[col]
constexpr int E_AUX  = 16;  // + aux[(row % aux_mod)*aux_ld + col] if valid

// 16-byte async copy global -> LDS (falls back to load+store if unavailable)
DEVI void async_cp16(const bf16* g, bf16* l)
{
#if ET_ASYNC
    __builtin_amdgcn_global_load_async_to_lds_b128(
        (__attribute__((address_space(1))) v4i*)(g),
        (__attribute__((address_space(3))) v4i*)(l), 0, 0);
#else
    *(v8bf*)l = *(const v8bf*)g;
#endif
}

DEVI void fence_async()
{
#if ET_ASYNC
#if __has_builtin(__builtin_amdgcn_s_wait_asynccnt)
    __builtin_amdgcn_s_wait_asynccnt(0);
#elif defined(__HIP_DEVICE_COMPILE__)
    asm volatile("s_wait_asynccnt 0x0" ::: "memory");
#endif
#endif
    __syncthreads();
}

// ---------------------------------------------------------------------------
// Generic batched bf16 GEMM:  C[M,N] = scale * A[M,K] x B[K,N]  (+ epilogue)
//   TA:  A(m,k) = A[k*lda + m]
//   TB:  B(k,n) = B[n*ldb + k]
//   BSW: B pre-swizzled: B[(((k/32)*NT + n/16)*32 + lane)*16 + idx]
// Block: 256 threads (8 waves), tile 128(M) x 128(N), K-step 32,
// double-buffered LDS. Wave w owns M-subtile w and all 8 N-subtiles.
// Batched via blockIdx.z = z1*bdiv + z2 with independent strides.
// ---------------------------------------------------------------------------
template<int EPI, bool TA, bool TB, bool BSW>
__global__ __launch_bounds__(256)
void gemm_k(const bf16* __restrict__ A, long lda, long bsA1, long bsA2,
            const bf16* __restrict__ B, long ldb, long bsB1, long bsB2, int NT,
            float* __restrict__ Cf, bf16* __restrict__ Cb,
            long ldc, long bsC1, long bsC2,
            int M, int N, int K, int bdiv,
            const float* __restrict__ bias,
            const float* __restrict__ aux, int aux_ld, int aux_mod, int aux_valid,
            const float* __restrict__ scale_ptr, float scale_imm)
{
    const int tid  = threadIdx.x;
    const int wave = tid >> 5;
    const int lane = tid & 31;

    const int bm = blockIdx.x * 128;
    const int bn = blockIdx.y * 128;

    const int z  = blockIdx.z;
    const int z1 = z / bdiv;
    const int z2 = z % bdiv;
    A += (long)z1 * bsA1 + (long)z2 * bsA2;
    B += (long)z1 * bsB1 + (long)z2 * bsB2;
    const long coff = (long)z1 * bsC1 + (long)z2 * bsC2;

    // double-buffered fragment-major LDS: [buf][subtile][lane][16 halves]
    __shared__ alignas(32) bf16 sA[2][8][32][16];
    __shared__ alignas(32) bf16 sB[2][8][32][16];

    v8f acc[8] = {};

    const int srow = lane & 15;          // row/col within subtile
    const int koff = (lane & 16) >> 1;   // lanes 16..31 -> +8 in K

    auto stageAB = [&](int k0, int buf) {
        // ---- A fragment for (subtile=wave, lane) ----
        {
            const int mg = bm + (wave << 4) + srow;
            bf16* lp = &sA[buf][wave][lane][0];
            if (!TA && (k0 + 32 <= K) && mg < M) {
                const bf16* p = A + (long)mg * lda + k0 + koff;
                async_cp16(p, lp);
                async_cp16(p + 16, lp + 8);
            } else {
                #pragma unroll
                for (int idx = 0; idx < 16; ++idx) {
                    int k  = ((idx & 8) << 1) + koff + (idx & 7);
                    int kg = k0 + k;
                    bf16 v = (bf16)0.0f;
                    if (mg < M && kg < K)
                        v = TA ? A[(long)kg * lda + mg] : A[(long)mg * lda + kg];
                    lp[idx] = v;
                }
            }
        }
        // ---- B fragment for (subtile=wave, lane) ----
        bf16* lp = &sB[buf][wave][lane][0];
        if (BSW) {
            const int tile = (bn >> 4) + wave;
            if (tile < NT) {
                const bf16* p = B + (((long)(k0 >> 5) * NT + tile) * 32 + lane) * 16;
                async_cp16(p, lp);
                async_cp16(p + 8, lp + 8);
            } else {
                v16bf zv = {};
                *(v16bf*)lp = zv;
            }
        } else {
            const int ng = bn + (wave << 4) + srow;
            if (TB && (k0 + 32 <= K) && ng < N) {
                const bf16* p = B + (long)ng * ldb + k0 + koff;
                async_cp16(p, lp);
                async_cp16(p + 16, lp + 8);
            } else {
                #pragma unroll
                for (int idx = 0; idx < 16; ++idx) {
                    int k  = ((idx & 8) << 1) + koff + (idx & 7);
                    int kg = k0 + k;
                    bf16 v = (bf16)0.0f;
                    if (ng < N && kg < K)
                        v = TB ? B[(long)ng * ldb + kg] : B[(long)kg * ldb + ng];
                    lp[idx] = v;
                }
            }
        }
    };

    const int nsteps = (K + 31) >> 5;
    stageAB(0, 0);
    fence_async();

    for (int i = 0; i < nsteps; ++i) {
        const int cur = i & 1;
        if (i + 1 < nsteps) stageAB((i + 1) << 5, cur ^ 1);

        const v16bf afr = *(const v16bf*)&sA[cur][wave][lane][0];
        v16bf bfr[8];
        #pragma unroll
        for (int t = 0; t < 8; ++t)
            bfr[t] = *(const v16bf*)&sB[cur][t][lane][0];
        #pragma unroll
        for (int t = 0; t < 8; ++t)
            acc[t] = __builtin_amdgcn_wmma_f32_16x16x32_bf16(
                false, afr, false, bfr[t], (short)0, acc[t], false, false);

        fence_async();
    }

    // ---- epilogue ----
    float s = scale_imm;
    if (scale_ptr) s *= scale_ptr[0];

    #pragma unroll
    for (int t = 0; t < 8; ++t) {
        #pragma unroll
        for (int r = 0; r < 8; ++r) {
            int row = bm + (wave << 4) + r + ((lane & 16) >> 1);
            int col = bn + (t << 4) + (lane & 15);
            if (row < M && col < N) {
                float v = acc[t][r] * s;
                if (EPI & E_BIAS) v += bias[col];
                if (EPI & E_AUX) {
                    int pr = row % aux_mod;
                    if (pr < aux_valid) v += aux[(long)pr * aux_ld + col];
                }
                long off = coff + (long)row * ldc + col;
                if (EPI & E_ACC)       Cf[off] += v;
                else if (EPI & E_F32)  Cf[off]  = v;
                if (EPI & E_BF16)      Cb[off]  = (bf16)v;
            }
        }
    }
}

// ---------------------------------------------------------------------------
// Convert f32 weights -> bf16 pre-swizzled into WMMA B-fragment order.
//   B(k,n) = trans ? src[n*srcld + k] : src[k*srcld + n]
//   dst[(((k/32 + kb)*NTtot + n/16 + tb)*32 + lane)*16 + idx]
//   lane = (n&15) | ((k&8)<<1),  idx = ((k&16)>>1) | (k&7)
// One call fills NTfill tiles starting at tile offset tb, K-block offset kb.
// Columns N..NTfill*16-1 are zero padding.
// ---------------------------------------------------------------------------
__global__ void conv_sw_k(const float* __restrict__ src, long srcld,
                          bf16* __restrict__ dst, int K, int N,
                          int NTfill, int NTtot, int trans, int kb, int tb)
{
    long i = (long)blockIdx.x * blockDim.x + threadIdx.x;
    long total = (long)K * NTfill * 16;
    if (i >= total) return;
    int npad = NTfill * 16;
    int k = (int)(i / npad);
    int n = (int)(i % npad);
    float v = 0.0f;
    if (n < N)
        v = trans ? src[(long)n * srcld + k] : src[(long)k * srcld + n];
    int tile = (n >> 4) + tb;
    int lane = (n & 15) | ((k & 8) << 1);
    int idx  = ((k & 16) >> 1) | (k & 7);
    dst[(((long)((k >> 5) + kb) * NTtot + tile) * 32 + lane) * 16 + idx] = (bf16)v;
}

// ---------------------------------------------------------------------------
// Patchify: x[B,3,224,224] -> patches bf16 [B, S_PAD, 768]
// ---------------------------------------------------------------------------
__global__ void patchify_k(const float* __restrict__ x, bf16* __restrict__ out)
{
    long i = (long)blockIdx.x * blockDim.x + threadIdx.x;
    if (i >= MT * (long)TKN) return;
    long b   = i / ((long)S_PAD * TKN);
    long rem = i % ((long)S_PAD * TKN);
    int tok = (int)(rem / TKN);
    int d   = (int)(rem % TKN);
    float v = 0.0f;
    if (tok >= 1 && tok <= 196) {
        int p  = tok - 1;
        int hh = p / 14, ww = p % 14;
        int c  = d % 3;
        int pp = d / 3;
        int p1 = pp / 16, p2 = pp % 16;
        v = x[(((b * 3 + c) * 224) + (hh * 16 + p1)) * 224 + (ww * 16 + p2)];
    }
    out[i] = (bf16)v;
}

// t[b,0,:] = cls + pos[0]
__global__ void fixcls_k(const float* __restrict__ cls, const float* __restrict__ pos,
                         float* __restrict__ t)
{
    int i = blockIdx.x * blockDim.x + threadIdx.x;
    if (i >= BATCH * TKN) return;
    int b = i / TKN, d = i % TKN;
    t[(long)b * S_PAD * TKN + d] = cls[d] + pos[d];
}

// ---------------------------------------------------------------------------
// Energy layer norm over 768 features; one 256-thread block per row.
// mod==0: all rows valid; else row valid iff (row % mod) < valid,
// invalid rows write zeros (keeps padded tokens inert downstream).
// ---------------------------------------------------------------------------
__global__ __launch_bounds__(256)
void ln_k(const float* __restrict__ X, long xstride,
          bf16* __restrict__ Y, long ystride,
          int mod, int valid,
          const float* __restrict__ gamma, const float* __restrict__ beta)
{
    long r = blockIdx.x;
    const float* x = X + r * xstride;
    bf16* y = Y + r * ystride;
    if (mod != 0 && (int)(r % mod) >= valid) {
        for (int i = threadIdx.x; i < TKN; i += 256) y[i] = (bf16)0.0f;
        return;
    }
    float vals[3], s = 0.0f, ss = 0.0f;
    #pragma unroll
    for (int j = 0; j < 3; ++j) {
        float v = x[threadIdx.x + j * 256];
        vals[j] = v; s += v; ss += v * v;
    }
    for (int o = 16; o > 0; o >>= 1) {
        s  += __shfl_xor(s,  o, 32);
        ss += __shfl_xor(ss, o, 32);
    }
    __shared__ float rs[8], rq[8];
    int wv = threadIdx.x >> 5, ln = threadIdx.x & 31;
    if (ln == 0) { rs[wv] = s; rq[wv] = ss; }
    __syncthreads();
    if (threadIdx.x == 0) {
        float a = 0, b2 = 0;
        for (int i = 0; i < 8; ++i) { a += rs[i]; b2 += rq[i]; }
        rs[0] = a; rq[0] = b2;
    }
    __syncthreads();
    float mean = rs[0] * (1.0f / TKN);
    float var  = rq[0] * (1.0f / TKN) - mean * mean;
    float inv  = rsqrtf(var + LN_EPS);
    float g0   = gamma[0];
    #pragma unroll
    for (int j = 0; j < 3; ++j) {
        int i = threadIdx.x + j * 256;
        y[i] = (bf16)(g0 * (vals[j] - mean) * inv + beta[i]);
    }
}

// ---------------------------------------------------------------------------
// Row softmax, in place on a bf16 matrix; one 256-thread block per row.
// Reads cols [0,valid), writes cols [0,width) (pad cols set to 0).
// Optional scalar scale from device memory (hop_beta) applied pre-softmax.
// ---------------------------------------------------------------------------
__global__ __launch_bounds__(256)
void softmax_k(bf16* __restrict__ D, long stride, int valid, int width,
               const float* __restrict__ scale_ptr)
{
    long r = blockIdx.x;
    bf16* d = D + r * stride;
    float sc = scale_ptr ? scale_ptr[0] : 1.0f;
    const int nit = (width + 255) >> 8;   // <= 12 (HID/256)
    float ev[12];

    float m = -3.0e38f;
    for (int j = 0; j < nit; ++j) {
        int i = threadIdx.x + (j << 8);
        float v = (i < valid) ? (float)d[i] * sc : -3.0e38f;
        ev[j] = v;
        m = fmaxf(m, v);
    }
    for (int o = 16; o > 0; o >>= 1) m = fmaxf(m, __shfl_xor(m, o, 32));
    __shared__ float rmax[8], rsum[8];
    int wv = threadIdx.x >> 5, ln = threadIdx.x & 31;
    if (ln == 0) rmax[wv] = m;
    __syncthreads();
    if (threadIdx.x == 0) {
        float a = rmax[0];
        for (int i = 1; i < 8; ++i) a = fmaxf(a, rmax[i]);
        rmax[0] = a;
    }
    __syncthreads();
    m = rmax[0];

    float s = 0.0f;
    for (int j = 0; j < nit; ++j) {
        int i = threadIdx.x + (j << 8);
        float e = (i < valid) ? __expf(ev[j] - m) : 0.0f;
        ev[j] = e; s += e;
    }
    for (int o = 16; o > 0; o >>= 1) s += __shfl_xor(s, o, 32);
    if (ln == 0) rsum[wv] = s;
    __syncthreads();
    if (threadIdx.x == 0) {
        float a = 0;
        for (int i = 0; i < 8; ++i) a += rsum[i];
        rsum[0] = a;
    }
    __syncthreads();
    float inv = 1.0f / rsum[0];
    for (int j = 0; j < nit; ++j) {
        int i = threadIdx.x + (j << 8);
        if (i < width) d[i] = (bf16)(ev[j] * inv);
    }
}

// ---------------------------------------------------------------------------
// Host-side orchestration
// ---------------------------------------------------------------------------
static inline dim3 g3(long M, long N, long z)
{
    return dim3((unsigned)((M + 127) / 128), (unsigned)((N + 127) / 128), (unsigned)z);
}

extern "C" void kernel_launch(void* const* d_in, const int* in_sizes, int n_in,
                              void* d_out, int out_size, void* d_ws, size_t ws_size,
                              hipStream_t stream)
{
    (void)in_sizes; (void)n_in; (void)out_size; (void)ws_size;

    const float* x        = (const float*)d_in[0];
    const float* w_enc    = (const float*)d_in[1];
    const float* b_enc    = (const float*)d_in[2];
    const float* cls      = (const float*)d_in[3];
    const float* pos      = (const float*)d_in[4];
    const float* n_gamma  = (const float*)d_in[5];
    const float* n_bias   = (const float*)d_in[6];
    const float* wq       = (const float*)d_in[7];
    const float* wk       = (const float*)d_in[8];
    const float* hop_w    = (const float*)d_in[9];
    const float* hop_beta = (const float*)d_in[10];
    const float* d_gamma  = (const float*)d_in[11];
    const float* d_bias   = (const float*)d_in[12];
    const float* w_dec    = (const float*)d_in[13];
    const float* b_dec    = (const float*)d_in[14];
    float* out = (float*)d_out;

    // ---- workspace carve-up ----
    char* base = (char*)d_ws;
    size_t off = 0;
    auto take = [&](size_t bytes) {
        size_t cur = off;
        off += (bytes + 255) & ~(size_t)255;
        return (void*)(base + cur);
    };
    float* t     = (float*)take(MT * (size_t)TKN * 4);
    bf16* patch  = (bf16*)take(MT * (size_t)TKN * 2);
    bf16* gbf    = (bf16*)take(MT * (size_t)TKN * 2);
    bf16* qk     = (bf16*)take(MT * (size_t)(2 * TKN) * 2);  // [q | k]
    bf16* dqdk   = (bf16*)take(MT * (size_t)(2 * TKN) * 2);  // [dq | dk]
    bf16* hbf    = (bf16*)take(MT * (size_t)HID * 2);
    bf16* Pbf    = (bf16*)take((size_t)BATCH * HEADS * S_PAD * S_PAD * 2);
    bf16* wencT  = (bf16*)take((size_t)TKN * TKN * 2);
    bf16* wqkT   = (bf16*)take((size_t)TKN * (2 * TKN) * 2); // K=768, N=1536
    bf16* wqkN   = (bf16*)take((size_t)(2 * TKN) * TKN * 2); // K=1536, N=768
    bf16* hopT   = (bf16*)take((size_t)TKN * HID * 2);
    bf16* hopN   = (bf16*)take((size_t)HID * TKN * 2);
    bf16* wdecT  = (bf16*)take((size_t)TKN * NCLS_P * 2);
    bf16* g0bf   = (bf16*)take((size_t)BATCH * TKN * 2);

    auto cgrid = [](long n) { return dim3((unsigned)((n + 255) / 256)); };
    const long W2 = (long)TKN * TKN;

    // ---- weight prep: bf16 + pre-swizzle into WMMA B-fragment order ----
    conv_sw_k<<<cgrid(W2), 256, 0, stream>>>(w_enc, TKN, wencT, TKN, TKN, 48, 48, 1, 0, 0);
    // W_qk^T : [768, 1536] = [Wq^T | Wk^T]
    conv_sw_k<<<cgrid(W2), 256, 0, stream>>>(wq, TKN, wqkT, TKN, TKN, 48, 96, 1, 0, 0);
    conv_sw_k<<<cgrid(W2), 256, 0, stream>>>(wk, TKN, wqkT, TKN, TKN, 48, 96, 1, 0, 48);
    // [Wq ; Wk] : [1536, 768]
    conv_sw_k<<<cgrid(W2), 256, 0, stream>>>(wq, TKN, wqkN, TKN, TKN, 48, 48, 0, 0, 0);
    conv_sw_k<<<cgrid(W2), 256, 0, stream>>>(wk, TKN, wqkN, TKN, TKN, 48, 48, 0, 24, 0);
    conv_sw_k<<<cgrid((long)TKN * HID), 256, 0, stream>>>(hop_w, TKN, hopT, TKN, HID, 192, 192, 1, 0, 0);
    conv_sw_k<<<cgrid((long)HID * TKN), 256, 0, stream>>>(hop_w, TKN, hopN, HID, TKN, 48, 48, 0, 0, 0);
    conv_sw_k<<<cgrid((long)TKN * NCLS_P), 256, 0, stream>>>(w_dec, TKN, wdecT, TKN, NCLS, 63, 63, 1, 0, 0);

    // ---- patchify + encoder: t = patch @ w_enc^T + b_enc + pos ----
    patchify_k<<<cgrid(MT * (long)TKN), 256, 0, stream>>>(x, patch);
    gemm_k<E_F32 | E_BIAS | E_AUX, false, false, true><<<g3(MT, TKN, 1), 256, 0, stream>>>(
        patch, TKN, 0, 0, wencT, 0, 0, 0, 48, t, nullptr, TKN, 0, 0,
        (int)MT, TKN, TKN, 1, b_enc, pos, TKN, S_PAD, S_TOK, nullptr, 1.0f);
    fixcls_k<<<cgrid((long)BATCH * TKN), 256, 0, stream>>>(cls, pos, t);

    const long QKLD  = 2 * TKN;                 // 1536
    const long QKB   = (long)S_PAD * QKLD;      // per-image stride in qk/dqdk
    const long PB    = (long)S_PAD * S_PAD;     // per-head stride in P

    // ---- 12 energy-descent steps ----
    for (int it = 0; it < K_STEPS; ++it) {
        // g = LN(t)  (bf16, pad rows zeroed)
        ln_k<<<dim3((unsigned)MT), 256, 0, stream>>>(t, TKN, gbf, TKN,
                                                     S_PAD, S_TOK, n_gamma, n_bias);
        // [q|k] = g @ [Wq^T | Wk^T]
        gemm_k<E_BF16, false, false, true><<<g3(MT, QKLD, 1), 256, 0, stream>>>(
            gbf, TKN, 0, 0, wqkT, 0, 0, 0, 96, nullptr, qk, QKLD, 0, 0,
            (int)MT, (int)QKLD, TKN, 1, nullptr, nullptr, 0, 1, 1, nullptr, 1.0f);

        // logits: P[b,h] = beta * q_bh @ k_bh^T   (batched NT, z = b*12+h)
        gemm_k<E_BF16, false, true, false><<<g3(S_PAD, S_PAD, BATCH * HEADS), 256, 0, stream>>>(
            qk,       QKLD, QKB, QKD,
            qk + TKN, QKLD, QKB, QKD, 0,
            nullptr, Pbf, S_PAD, (long)HEADS * PB, PB,
            S_PAD, S_PAD, QKD, HEADS, nullptr, nullptr, 0, 1, 1, nullptr, ATTN_BETA);
        // softmax over keys (valid 197, pad cols -> 0)
        softmax_k<<<dim3((unsigned)((long)BATCH * HEADS * S_PAD)), 256, 0, stream>>>(
            Pbf, S_PAD, S_TOK, S_PAD, nullptr);

        // dq = P @ k ; dk = P^T @ q   (batched, written side by side)
        gemm_k<E_BF16, false, false, false><<<g3(S_PAD, QKD, BATCH * HEADS), 256, 0, stream>>>(
            Pbf, S_PAD, (long)HEADS * PB, PB,
            qk + TKN, QKLD, QKB, QKD, 0,
            nullptr, dqdk, QKLD, QKB, QKD,
            S_PAD, QKD, S_PAD, HEADS, nullptr, nullptr, 0, 1, 1, nullptr, 1.0f);
        gemm_k<E_BF16, true, false, false><<<g3(S_PAD, QKD, BATCH * HEADS), 256, 0, stream>>>(
            Pbf, S_PAD, (long)HEADS * PB, PB,
            qk, QKLD, QKB, QKD, 0,
            nullptr, dqdk + TKN, QKLD, QKB, QKD,
            S_PAD, QKD, S_PAD, HEADS, nullptr, nullptr, 0, 1, 1, nullptr, 1.0f);

        // t += [dq|dk] @ [Wq ; Wk]   (minus-grad folded into signs)
        gemm_k<E_ACC, false, false, true><<<g3(MT, TKN, 1), 256, 0, stream>>>(
            dqdk, QKLD, 0, 0, wqkN, 0, 0, 0, 48, t, nullptr, TKN, 0, 0,
            (int)MT, TKN, (int)QKLD, 1, nullptr, nullptr, 0, 1, 1, nullptr, 1.0f);

        // Hopfield: h = g @ hop_w^T ; sm = softmax(beta_h*h) ; t += sm @ hop_w
        gemm_k<E_BF16, false, false, true><<<g3(MT, HID, 1), 256, 0, stream>>>(
            gbf, TKN, 0, 0, hopT, 0, 0, 0, 192, nullptr, hbf, HID, 0, 0,
            (int)MT, HID, TKN, 1, nullptr, nullptr, 0, 1, 1, nullptr, 1.0f);
        softmax_k<<<dim3((unsigned)MT), 256, 0, stream>>>(hbf, HID, HID, HID, hop_beta);
        gemm_k<E_ACC, false, false, true><<<g3(MT, TKN, 1), 256, 0, stream>>>(
            hbf, HID, 0, 0, hopN, 0, 0, 0, 48, t, nullptr, TKN, 0, 0,
            (int)MT, TKN, HID, 1, nullptr, nullptr, 0, 1, 1, nullptr, 1.0f);
    }

    // ---- decode: LN(t[:,0]) @ w_dec^T + b_dec ----
    ln_k<<<dim3((unsigned)BATCH), 256, 0, stream>>>(
        t, (long)S_PAD * TKN, g0bf, TKN, 0, 0, d_gamma, d_bias);
    gemm_k<E_F32 | E_BIAS, false, false, true><<<g3(BATCH, NCLS, 1), 256, 0, stream>>>(
        g0bf, TKN, 0, 0, wdecT, 0, 0, 0, 63, out, nullptr, NCLS, 0, 0,
        BATCH, NCLS, TKN, 1, b_dec, nullptr, 0, 1, 1, nullptr, 1.0f);
}
